// GCN_48378511622248
// MI455X (gfx1250) — matrix operand order repaired and checked by reading, not compiled, not run on hardware
//
#include <hip/hip_runtime.h>
#include <hip/hip_bf16.h>

typedef __attribute__((ext_vector_type(16))) __bf16 v16bf;
typedef __attribute__((ext_vector_type(8)))  float  v8f;

// ---------------------------------------------------------------------------
// Dense GEMM  C[M,N] = A[M,K] @ B[K,N]   (fp32 in, bf16 WMMA, fp32 accum)
// Block = 256 threads = 8 waves. Wave w handles rows [ (bid*8+w)*16 , +16 ).
// Each wave keeps all N/16 accumulator tiles live so A is streamed once.
// B k-slab (32 x N) staged in LDS as bf16, transposed [n][k] so each lane's
// fragment is two runs of 8 contiguous halves (ds_load_b128-friendly).
// ---------------------------------------------------------------------------
template<int NCOLS>
__global__ __launch_bounds__(256)
void gemm_bf16_wmma(const float* __restrict__ A, const float* __restrict__ B,
                    float* __restrict__ C, int M, int K)
{
    constexpr int NT = NCOLS / 16;
    __shared__ __bf16 Bs[NCOLS * 32];          // [n][k] transposed slab

    const int tid  = threadIdx.x;
    const int wave = tid >> 5;
    const int lane = tid & 31;
    const int half = lane >> 4;                // 0: lanes 0-15, 1: lanes 16-31
    const int ln   = lane & 15;
    const int rowBase = (blockIdx.x * 8 + wave) * 16;
    const bool active = rowBase < M;           // M is a multiple of 16

    v8f acc[NT] = {};

    const float* arow = A + (size_t)(active ? (rowBase + ln) : ln) * K;

    for (int kb = 0; kb < K; kb += 32) {
        __syncthreads();
        // stage W[kb:kb+32, 0:N] into LDS as bf16, transposed
        for (int i = tid; i < NCOLS * 32; i += 256) {
            int kk = i / NCOLS, nn = i % NCOLS;
            Bs[nn * 32 + kk] = (__bf16)B[(size_t)(kb + kk) * NCOLS + nn];
        }
        __syncthreads();

        // ----- A fragment (ISA 16-bit A 16x32 layout) -----
        // lanes 0-15 : elems 0-7 => K=kb+0..7,  elems 8-15 => K=kb+16..23
        // lanes 16-31: elems 0-7 => K=kb+8..15, elems 8-15 => K=kb+24..31
        float4 f0 = {0,0,0,0}, f1 = f0, f2 = f0, f3 = f0;
        if (active) {
            const int k0 = kb + half * 8;
            const int k1 = kb + 16 + half * 8;
            f0 = *(const float4*)(arow + k0);
            f1 = *(const float4*)(arow + k0 + 4);
            f2 = *(const float4*)(arow + k1);
            f3 = *(const float4*)(arow + k1 + 4);
        }
        v16bf af;
        af[0]  = (__bf16)f0.x; af[1]  = (__bf16)f0.y;
        af[2]  = (__bf16)f0.z; af[3]  = (__bf16)f0.w;
        af[4]  = (__bf16)f1.x; af[5]  = (__bf16)f1.y;
        af[6]  = (__bf16)f1.z; af[7]  = (__bf16)f1.w;
        af[8]  = (__bf16)f2.x; af[9]  = (__bf16)f2.y;
        af[10] = (__bf16)f2.z; af[11] = (__bf16)f2.w;
        af[12] = (__bf16)f3.x; af[13] = (__bf16)f3.y;
        af[14] = (__bf16)f3.z; af[15] = (__bf16)f3.w;

        // ----- B fragments + WMMA over all column tiles -----
        // B layout: lane = N; lanes 0-15 hold K=kb+0..7,16..23, lanes 16-31 K=kb+8..15,24..31
        #pragma unroll
        for (int t = 0; t < NT; ++t) {
            const __bf16* bp = &Bs[(t * 16 + ln) * 32 + half * 8];
            v16bf bf;
            #pragma unroll
            for (int e = 0; e < 16; ++e) bf[e] = bp[e < 8 ? e : e + 8];
            acc[t] = __builtin_amdgcn_wmma_f32_16x16x32_bf16(
                false, af, false, bf, (short)0, acc[t], false, false);
        }
    }

    if (active) {
        // C/D layout: lane = N; vgpr e => M = half*8 + e
        #pragma unroll
        for (int t = 0; t < NT; ++t) {
            const int col = t * 16 + ln;
            #pragma unroll
            for (int e = 0; e < 8; ++e) {
                const int row = rowBase + half * 8 + e;
                C[(size_t)row * NCOLS + col] = acc[t][e];
            }
        }
    }
}

// ---------------------------------------------------------------------------
// Zero a float buffer (graph-replay-safe re-init of atomic targets)
// ---------------------------------------------------------------------------
__global__ void zero_f32(float* __restrict__ p, long long n)
{
    long long i = (long long)blockIdx.x * blockDim.x + threadIdx.x;
    if (i < n) p[i] = 0.0f;
}

// ---------------------------------------------------------------------------
// Edge-parallel scatter-aggregate: agg[dst[e], :] += adj[e] * sup[src[e], :]
// One thread per (edge, 4-feature chunk); fshift = log2(F/4).
// support fits in 192MB L2 -> gather + fp32 atomics are L2-resident.
// ---------------------------------------------------------------------------
__global__ void scatter_agg(const float* __restrict__ sup,
                            const int* __restrict__ src,
                            const int* __restrict__ dst,
                            const float* __restrict__ val,
                            float* __restrict__ agg,
                            long long total, int fshift)
{
    long long tid = (long long)blockIdx.x * blockDim.x + threadIdx.x;
    if (tid >= total) return;
    const int e = (int)(tid >> fshift);
    const int c = (int)(tid & ((1 << fshift) - 1));

    const float a = val[e];
    const int   s = src[e];
    const int   d = dst[e];

    const float4 sv = ((const float4*)(sup + ((size_t)s << (fshift + 2))))[c];
    float* ap = agg + ((size_t)d << (fshift + 2)) + (size_t)c * 4;
    unsafeAtomicAdd(ap + 0, sv.x * a);
    unsafeAtomicAdd(ap + 1, sv.y * a);
    unsafeAtomicAdd(ap + 2, sv.z * a);
    unsafeAtomicAdd(ap + 3, sv.w * a);
}

// ---------------------------------------------------------------------------
// In-place bias + ReLU:  h[i] = max(h[i] + b[i & (F-1)], 0)
// ---------------------------------------------------------------------------
__global__ void bias_relu(float* __restrict__ h, const float* __restrict__ b,
                          long long total, int fmask)
{
    long long i = (long long)blockIdx.x * blockDim.x + threadIdx.x;
    if (i >= total) return;
    float v = h[i] + b[i & fmask];
    h[i] = v > 0.0f ? v : 0.0f;
}

extern "C" void kernel_launch(void* const* d_in, const int* in_sizes, int n_in,
                              void* d_out, int out_size, void* d_ws, size_t ws_size,
                              hipStream_t stream)
{
    const float* x        = (const float*)d_in[0];
    const int*   edge_src = (const int*)  d_in[1];
    const int*   edge_dst = (const int*)  d_in[2];
    const float* adj_vals = (const float*)d_in[3];
    const float* W1       = (const float*)d_in[4];
    const float* b1       = (const float*)d_in[5];
    const float* W2       = (const float*)d_in[6];
    const float* b2       = (const float*)d_in[7];
    float* out = (float*)d_out;

    const int M = in_sizes[0] / 512;   // 100000 nodes (multiple of 16)
    const int E = in_sizes[1];         // 3.2M edges
    constexpr int NH1 = 256, NH2 = 128, NF = 512;

    float* support1 = (float*)d_ws;                       // M*256
    float* h1       = support1 + (size_t)M * NH1;         // M*256 (agg1 in place)
    float* support2 = h1       + (size_t)M * NH1;         // M*128

    const int gemmBlocks = ((M + 15) / 16 + 7) / 8;
    const long long n1 = (long long)M * NH1;
    const long long n2 = (long long)M * NH2;
    const long long sc1 = (long long)E * (NH1 / 4);
    const long long sc2 = (long long)E * (NH2 / 4);

    // ---- Layer 1 ----
    gemm_bf16_wmma<NH1><<<gemmBlocks, 256, 0, stream>>>(x, W1, support1, M, NF);
    zero_f32<<<(int)((n1 + 255) / 256), 256, 0, stream>>>(h1, n1);
    scatter_agg<<<(int)((sc1 + 255) / 256), 256, 0, stream>>>(
        support1, edge_src, edge_dst, adj_vals, h1, sc1, 6 /* log2(256/4) */);
    bias_relu<<<(int)((n1 + 255) / 256), 256, 0, stream>>>(h1, b1, n1, NH1 - 1);

    // ---- Layer 2 ----
    gemm_bf16_wmma<NH2><<<gemmBlocks, 256, 0, stream>>>(h1, W2, support2, M, NH1);
    zero_f32<<<(int)((n2 + 255) / 256), 256, 0, stream>>>(out, n2);
    scatter_agg<<<(int)((sc2 + 255) / 256), 256, 0, stream>>>(
        support2, edge_src, edge_dst, adj_vals, out, sc2, 5 /* log2(128/4) */);
    bias_relu<<<(int)((n2 + 255) / 256), 256, 0, stream>>>(out, b2, n2, NH2 - 1);
}
